// MultiHeadAttention_11321533792894
// MI455X (gfx1250) — matrix operand order repaired and checked by reading, not compiled, not run on hardware
//
#include <hip/hip_runtime.h>

// CDNA5 (gfx1250) wave32 WMMA types
typedef __attribute__((ext_vector_type(16))) __bf16 v16bf;
typedef __attribute__((ext_vector_type(8)))  float  v8f;

// ---------------------------------------------------------------------------
// Helpers
// ---------------------------------------------------------------------------
__device__ inline unsigned pack_bf16x2(float lo, float hi) {
    unsigned short a = __builtin_bit_cast(unsigned short, (__bf16)lo);
    unsigned short b = __builtin_bit_cast(unsigned short, (__bf16)hi);
    return (unsigned)a | ((unsigned)b << 16);
}

// cross-half (lane ^ 16) exchange inside a wave32
__device__ inline unsigned xhalf_u32(unsigned x) {
    int lane = (int)(threadIdx.x & 31u);
    return (unsigned)__builtin_amdgcn_ds_bpermute((lane ^ 16) << 2, (int)x);
}
__device__ inline float xhalf_f32(float x) {
    return __builtin_bit_cast(float, (int)xhalf_u32(__builtin_bit_cast(unsigned, x)));
}

union Frag {
    uint4    q[2];
    unsigned u[8];
    v16bf    v;
};

__device__ inline v8f wmma_bf16(const Frag& a, const Frag& b, v8f c) {
    return __builtin_amdgcn_wmma_f32_16x16x32_bf16(false, a.v, false, b.v,
                                                   (short)0, c, false, false);
}

__device__ inline v8f vzero8() { return (v8f){0.f,0.f,0.f,0.f,0.f,0.f,0.f,0.f}; }

// Async global -> LDS DMA (128-bit per lane), tracked by ASYNCcnt.
__device__ inline void async_g2l_b128(unsigned lds, const void* g) {
    asm volatile("global_load_async_to_lds_b128 %0, %1, off"
                 :: "v"(lds), "v"(g) : "memory");
}
__device__ inline unsigned lds_off(const void* p) {
    return (unsigned)(uintptr_t)p;  // LDS aperture: addr[31:0] = LDS offset
}

// B-operand loader straight from global (used for Q in attention):
// lane<16 holds K=k..k+15 of column n=lane (32 contiguous bytes/lane).
__device__ inline Frag load_b_global(const __bf16* __restrict__ base, int ld, int col0, int k) {
    int lane = threadIdx.x & 31;
    const __bf16* p = base + (size_t)(col0 + (lane & 15)) * ld + k + (lane >> 4) * 16;
    Frag f;
    f.q[0] = *reinterpret_cast<const uint4*>(p);
    f.q[1] = *reinterpret_cast<const uint4*>(p + 8);
    return f;
}

// ---------------------------------------------------------------------------
// f32 -> bf16 cast (vectorized x4)
// ---------------------------------------------------------------------------
__global__ __launch_bounds__(256) void cast_bf16_kernel(const float* __restrict__ in,
                                                        __bf16* __restrict__ out, int n4) {
    int i = blockIdx.x * blockDim.x + threadIdx.x;
    if (i < n4) {
        float4 f = reinterpret_cast<const float4*>(in)[i];
        uint2 o;
        o.x = pack_bf16x2(f.x, f.y);
        o.y = pack_bf16x2(f.z, f.w);
        reinterpret_cast<uint2*>(out)[i] = o;
    }
}

// ---------------------------------------------------------------------------
// Staged GEMM core: block tile 128M x 256N, 8 waves (2x4), wave tile 64x64.
// K-depth 64 per LDS buffer (two WMMA k-steps of 32), double-buffered async
// DMA. Rows padded to 72 elements (144 B = 36 banks -> conflict-free frags).
// 32 WMMAs per barrier pair; 12 async ops/wave/tile.
// ---------------------------------------------------------------------------
#define LDT 72  // padded row stride (elements) for 64-wide K tiles

__device__ __forceinline__ void gemm_core_bf16(const __bf16* __restrict__ X,
                                               const __bf16* __restrict__ W,
                                               v8f acc[4][4]) {
    const int D = 1024;
    __shared__ alignas(16) __bf16 sA[2 * 128 * LDT];
    __shared__ alignas(16) __bf16 sB[2 * 256 * LDT];

    int tid  = threadIdx.x;
    int lane = tid & 31, wave = tid >> 5;
    int mloc = (wave & 1) * 64, nloc = (wave >> 1) * 64;
    int l15  = lane & 15, hf = lane >> 4;
    size_t mblk = (size_t)blockIdx.y * 128, nblk = (size_t)blockIdx.x * 256;

    unsigned baseA = lds_off(&sA[0]);
    unsigned baseB = lds_off(&sB[0]);

    auto issue = [&](int buf, int k) {
        unsigned a0 = baseA + (unsigned)buf * (128 * LDT * 2);
        unsigned b0 = baseB + (unsigned)buf * (256 * LDT * 2);
        // A tile 128x64: 1024 b128 chunks (8 per row), 4 per thread
#pragma unroll
        for (int i = 0; i < 4; i++) {
            int c = tid + i * 256;
            int row = c >> 3, cc = c & 7;
            async_g2l_b128(a0 + (unsigned)(row * (LDT * 2) + cc * 16),
                           X + (mblk + row) * D + k + cc * 8);
        }
        // B tile 256x64: 2048 b128 chunks (8 per row), 8 per thread
#pragma unroll
        for (int i = 0; i < 8; i++) {
            int c = tid + i * 256;
            int row = c >> 3, cc = c & 7;
            async_g2l_b128(b0 + (unsigned)(row * (LDT * 2) + cc * 16),
                           W + (nblk + row) * D + k + cc * 8);
        }
    };

    issue(0, 0);
    const int NK = D / 64;  // 16
    for (int ks = 0; ks < NK; ks++) {
        int buf = ks & 1;
        if (ks + 1 < NK) {
            issue(1 - buf, (ks + 1) * 64);
            asm volatile("s_wait_asynccnt 0xc" ::: "memory");  // prev tile landed
        } else {
            asm volatile("s_wait_asynccnt 0x0" ::: "memory");
        }
        __syncthreads();

        const __bf16* As = &sA[buf * 128 * LDT];
        const __bf16* Bs = &sB[buf * 256 * LDT];
#pragma unroll
        for (int kk = 0; kk < 64; kk += 32) {
            Frag a[4], b[4];
#pragma unroll
            for (int i = 0; i < 4; i++) {
                const __bf16* p = As + (mloc + i * 16 + l15) * LDT + kk + hf * 16;
                a[i].q[0] = *reinterpret_cast<const uint4*>(p);
                a[i].q[1] = *reinterpret_cast<const uint4*>(p + 8);
            }
#pragma unroll
            for (int j = 0; j < 4; j++) {
                const __bf16* p = Bs + (nloc + j * 16 + l15) * LDT + kk + hf * 16;
                b[j].q[0] = *reinterpret_cast<const uint4*>(p);
                b[j].q[1] = *reinterpret_cast<const uint4*>(p + 8);
            }
#pragma unroll
            for (int i = 0; i < 4; i++)
#pragma unroll
                for (int j = 0; j < 4; j++)
                    acc[i][j] = wmma_bf16(a[i], b[j], acc[i][j]);
        }
        __syncthreads();  // everyone done reading buf before it is re-DMA'd
    }
}

// ---------------------------------------------------------------------------
// Projection GEMM: Y = X @ W^T + bias. vmode 0/1 -> head-split [B,H,S,Dh];
// vmode 2 -> V^T [B,H,Dh,S] (bf16, packed b32 stores along s).
// ---------------------------------------------------------------------------
__global__ __launch_bounds__(256) void proj_gemm_kernel(const __bf16* __restrict__ X,
                                                        const __bf16* __restrict__ W,
                                                        const float*  __restrict__ bias,
                                                        __bf16* __restrict__ out, int vmode) {
    const int S = 2048, H = 16;
    int lane = threadIdx.x & 31;
    int wave = threadIdx.x >> 5;
    int m0 = blockIdx.y * 128 + (wave & 1) * 64;
    int n0 = blockIdx.x * 256 + (wave >> 1) * 64;
    int hf = lane >> 4, l15 = lane & 15;

    v8f acc[4][4];
#pragma unroll
    for (int i = 0; i < 4; i++)
#pragma unroll
        for (int j = 0; j < 4; j++) acc[i][j] = vzero8();

    gemm_core_bf16(X, W, acc);

    int bidx = m0 >> 11;  // / 2048
#pragma unroll
    for (int j = 0; j < 4; j++) {
        int col = n0 + j * 16 + l15;
        float bb = bias[col];
        int h = col >> 6, d = col & 63;
#pragma unroll
        for (int i = 0; i < 4; i++) {
            int sbase = (m0 & 2047) + i * 16 + hf * 8;  // C-layout: M = r + 8*half
            if (vmode == 2) {
                __bf16* op = out + ((size_t)(bidx * H + h) * 64 + d) * S + sbase;
#pragma unroll
                for (int r = 0; r < 8; r += 2) {
                    unsigned pk = pack_bf16x2(acc[i][j][r] + bb, acc[i][j][r + 1] + bb);
                    *reinterpret_cast<unsigned*>(op + r) = pk;
                }
            } else {
#pragma unroll
                for (int r = 0; r < 8; r++) {
                    out[((size_t)(bidx * H + h) * S + sbase + r) * 64 + d] =
                        (__bf16)(acc[i][j][r] + bb);
                }
            }
        }
    }
}

// ---------------------------------------------------------------------------
// Output GEMM: out(f32) = AOut(bf16) @ Wo^T + bo
// ---------------------------------------------------------------------------
__global__ __launch_bounds__(256) void out_gemm_kernel(const __bf16* __restrict__ X,
                                                       const __bf16* __restrict__ W,
                                                       const float*  __restrict__ bias,
                                                       float* __restrict__ out) {
    const int D = 1024;
    int lane = threadIdx.x & 31;
    int wave = threadIdx.x >> 5;
    int m0 = blockIdx.y * 128 + (wave & 1) * 64;
    int n0 = blockIdx.x * 256 + (wave >> 1) * 64;
    int hf = lane >> 4, l15 = lane & 15;

    v8f acc[4][4];
#pragma unroll
    for (int i = 0; i < 4; i++)
#pragma unroll
        for (int j = 0; j < 4; j++) acc[i][j] = vzero8();

    gemm_core_bf16(X, W, acc);

#pragma unroll
    for (int j = 0; j < 4; j++) {
        int col = n0 + j * 16 + l15;
        float bb = bias[col];
#pragma unroll
        for (int i = 0; i < 4; i++) {
            int mbase = m0 + i * 16 + hf * 8;
#pragma unroll
            for (int r = 0; r < 8; r++)
                out[(size_t)(mbase + r) * D + col] = acc[i][j][r] + bb;
        }
    }
}

// ---------------------------------------------------------------------------
// Flash attention. Block = 4 waves, one 64-query tile per wave, all waves
// share (b,h): 64-key K (64x64) and V^T (64x64) tiles are async-DMA'd into
// LDS (double-buffered) and consumed as two 32-key softmax sub-blocks.
// Scores computed transposed (S^T = K*Q^T): softmax reduces in-register,
// exp(P) converts to the PV WMMA B-operand with packs + ds_bpermute.
// ---------------------------------------------------------------------------
__global__ __launch_bounds__(128) void attn_kernel(const __bf16* __restrict__ Qh,
                                                   const __bf16* __restrict__ Kh,
                                                   const __bf16* __restrict__ Vt,
                                                   __bf16* __restrict__ AOut) {
    const int S = 2048, Dh = 64, H = 16, D = 1024;
    __shared__ alignas(16) __bf16 sK[2 * 64 * LDT];
    __shared__ alignas(16) __bf16 sV[2 * 64 * LDT];

    int tid  = threadIdx.x;
    int lane = tid & 31;
    int wave = tid >> 5;
    int b = blockIdx.z, h = blockIdx.y;
    int m0 = (blockIdx.x * 4 + wave) * 64;
    int hf = lane >> 4, l15 = lane & 15;

    const __bf16* Qb = Qh + (size_t)(b * H + h) * S * Dh;
    const __bf16* Kb = Kh + (size_t)(b * H + h) * S * Dh;
    const __bf16* Vb = Vt + (size_t)(b * H + h) * Dh * S;

    unsigned baseK = lds_off(&sK[0]);
    unsigned baseV = lds_off(&sV[0]);

    auto issue = [&](int buf, int t0) {
        unsigned k0 = baseK + (unsigned)buf * (64 * LDT * 2);
        unsigned v0 = baseV + (unsigned)buf * (64 * LDT * 2);
        // K tile 64x64: 512 b128 chunks (8/row), 4 per thread
#pragma unroll
        for (int i = 0; i < 4; i++) {
            int c = tid + i * 128;
            int row = c >> 3, cc = c & 7;
            async_g2l_b128(k0 + (unsigned)(row * (LDT * 2) + cc * 16),
                           Kb + (size_t)(t0 + row) * Dh + cc * 8);
        }
        // V^T tile 64x64 (d rows, t cols): 512 b128 chunks, 4 per thread
#pragma unroll
        for (int i = 0; i < 4; i++) {
            int c = tid + i * 128;
            int row = c >> 3, cc = c & 7;
            async_g2l_b128(v0 + (unsigned)(row * (LDT * 2) + cc * 16),
                           Vb + (size_t)row * S + t0 + cc * 8);
        }
    };

    // Q as B-operand, register-resident: 4 query tiles x 2 K-steps over Dh=64
    Frag bq[4][2];
#pragma unroll
    for (int mj = 0; mj < 4; mj++)
#pragma unroll
        for (int ks = 0; ks < 2; ks++) bq[mj][ks] = load_b_global(Qb, Dh, m0 + mj * 16, ks * 32);

    v8f ot[4][4];  // O^T accumulators [d-tile][q-tile]
#pragma unroll
    for (int i = 0; i < 4; i++)
#pragma unroll
        for (int j = 0; j < 4; j++) ot[i][j] = vzero8();

    float mrun[4], lrun[4];
#pragma unroll
    for (int mj = 0; mj < 4; mj++) { mrun[mj] = -3.0e38f; lrun[mj] = 0.f; }

    const float CEXP = 0.18033688011112042f;  // (1/sqrt(Dh)) * log2(e)

    issue(0, 0);
    for (int t0 = 0; t0 < S; t0 += 64) {
        int buf = (t0 >> 6) & 1;
        if (t0 + 64 < S) {
            issue(1 - buf, t0 + 64);
            asm volatile("s_wait_asynccnt 0x8" ::: "memory");
        } else {
            asm volatile("s_wait_asynccnt 0x0" ::: "memory");
        }
        __syncthreads();

        const __bf16* Ks = &sK[buf * 64 * LDT];
        const __bf16* Vs = &sV[buf * 64 * LDT];

#pragma unroll
        for (int ts = 0; ts < 64; ts += 32) {
            // ---- scores^T tiles: sc[ti][mj] (keys on VGPR rows, queries on lanes)
            v8f sc[2][4];
#pragma unroll
            for (int ti = 0; ti < 2; ti++)
#pragma unroll
                for (int mj = 0; mj < 4; mj++) sc[ti][mj] = vzero8();
#pragma unroll
            for (int ks = 0; ks < 2; ks++) {
                Frag ak[2];
#pragma unroll
                for (int ti = 0; ti < 2; ti++) {
                    const __bf16* p = Ks + (ts + ti * 16 + l15) * LDT + ks * 32 + hf * 16;
                    ak[ti].q[0] = *reinterpret_cast<const uint4*>(p);
                    ak[ti].q[1] = *reinterpret_cast<const uint4*>(p + 8);
                }
#pragma unroll
                for (int mj = 0; mj < 4; mj++) {
                    sc[0][mj] = wmma_bf16(ak[0], bq[mj][ks], sc[0][mj]);
                    sc[1][mj] = wmma_bf16(ak[1], bq[mj][ks], sc[1][mj]);
                }
            }

            // V^T A-operand fragments for this 32-key sub-chunk
            Frag av[4];
#pragma unroll
            for (int di = 0; di < 4; di++) {
                const __bf16* p = Vs + (di * 16 + l15) * LDT + ts + hf * 16;
                av[di].q[0] = *reinterpret_cast<const uint4*>(p);
                av[di].q[1] = *reinterpret_cast<const uint4*>(p + 8);
            }

#pragma unroll
            for (int mj = 0; mj < 4; mj++) {
                float mx = -3.0e38f;
#pragma unroll
                for (int ti = 0; ti < 2; ti++)
#pragma unroll
                    for (int r = 0; r < 8; r++) mx = fmaxf(mx, sc[ti][mj][r]);
                mx = fmaxf(mx, xhalf_f32(mx));
                float mnew  = fmaxf(mrun[mj], mx);
                float alpha = exp2f((mrun[mj] - mnew) * CEXP);
                float ls = 0.f;
#pragma unroll
                for (int ti = 0; ti < 2; ti++)
#pragma unroll
                    for (int r = 0; r < 8; r++) {
                        float p = exp2f((sc[ti][mj][r] - mnew) * CEXP);
                        sc[ti][mj][r] = p;
                        ls += p;
                    }
                ls += xhalf_f32(ls);
                lrun[mj] = lrun[mj] * alpha + ls;
                mrun[mj] = mnew;
#pragma unroll
                for (int di = 0; di < 4; di++)
#pragma unroll
                    for (int r = 0; r < 8; r++) ot[di][mj][r] *= alpha;

                // C-layout exp(P) -> B-operand: pack key pairs, swap halves
                unsigned p0[4], p1[4];
#pragma unroll
                for (int qq = 0; qq < 4; qq++) {
                    p0[qq] = pack_bf16x2(sc[0][mj][2 * qq], sc[0][mj][2 * qq + 1]);
                    p1[qq] = pack_bf16x2(sc[1][mj][2 * qq], sc[1][mj][2 * qq + 1]);
                }
                Frag bp;
#pragma unroll
                for (int qq = 0; qq < 4; qq++) {
                    unsigned s0 = xhalf_u32(p0[qq]);
                    unsigned s1 = xhalf_u32(p1[qq]);
                    bp.u[qq]     = hf ? s1 : p0[qq];
                    bp.u[qq + 4] = hf ? p1[qq] : s0;
                }
#pragma unroll
                for (int di = 0; di < 4; di++) ot[di][mj] = wmma_bf16(av[di], bp, ot[di][mj]);
            }
        }
        __syncthreads();
    }

    // normalize by l and store head-concat bf16 [B, S, H*Dh]
#pragma unroll
    for (int mj = 0; mj < 4; mj++) {
        float inv = 1.0f / lrun[mj];
        int srow = m0 + mj * 16 + l15;
#pragma unroll
        for (int di = 0; di < 4; di++) {
#pragma unroll
            for (int r = 0; r < 8; r++) {
                int d = di * 16 + hf * 8 + r;
                AOut[(size_t)(b * S + srow) * D + h * 64 + d] = (__bf16)(ot[di][mj][r] * inv);
            }
        }
    }
}

// ---------------------------------------------------------------------------
// Host launcher
// ---------------------------------------------------------------------------
extern "C" void kernel_launch(void* const* d_in, const int* in_sizes, int n_in,
                              void* d_out, int out_size, void* d_ws, size_t ws_size,
                              hipStream_t stream) {
    const int B = 4, S = 2048, D = 1024, H = 16;
    const size_t M = (size_t)B * S;  // 8192

    const float* q  = (const float*)d_in[0];
    const float* k  = (const float*)d_in[1];
    const float* v  = (const float*)d_in[2];
    const float* Wq = (const float*)d_in[3];
    const float* bq = (const float*)d_in[4];
    const float* Wk = (const float*)d_in[5];
    const float* bk = (const float*)d_in[6];
    const float* Wv = (const float*)d_in[7];
    const float* bv = (const float*)d_in[8];
    const float* Wo = (const float*)d_in[9];
    const float* bo = (const float*)d_in[10];
    float* out = (float*)d_out;

    char* ws = (char*)d_ws;
    size_t off = 0;
    auto alloc = [&](size_t bytes) {
        char* p = ws + off;
        off += (bytes + 255) & ~(size_t)255;
        return p;
    };
    __bf16* qb  = (__bf16*)alloc(M * D * 2);
    __bf16* kb  = (__bf16*)alloc(M * D * 2);
    __bf16* vb  = (__bf16*)alloc(M * D * 2);
    __bf16* Wqb = (__bf16*)alloc((size_t)D * D * 2);
    __bf16* Wkb = (__bf16*)alloc((size_t)D * D * 2);
    __bf16* Wvb = (__bf16*)alloc((size_t)D * D * 2);
    __bf16* Wob = (__bf16*)alloc((size_t)D * D * 2);
    __bf16* Qh  = (__bf16*)alloc(M * D * 2);  // [B,H,S,Dh]
    __bf16* Kh  = (__bf16*)alloc(M * D * 2);  // [B,H,S,Dh]
    __bf16* Vt  = (__bf16*)alloc(M * D * 2);  // [B,H,Dh,S]
    __bf16* AO  = (__bf16*)alloc(M * D * 2);  // [B,S,D] head-concat
    (void)ws_size;

    // 1) casts
    {
        int n4x = (int)(M * D / 4);
        int n4w = (int)((size_t)D * D / 4);
        cast_bf16_kernel<<<(n4x + 255) / 256, 256, 0, stream>>>(q, qb, n4x);
        cast_bf16_kernel<<<(n4x + 255) / 256, 256, 0, stream>>>(k, kb, n4x);
        cast_bf16_kernel<<<(n4x + 255) / 256, 256, 0, stream>>>(v, vb, n4x);
        cast_bf16_kernel<<<(n4w + 255) / 256, 256, 0, stream>>>(Wq, Wqb, n4w);
        cast_bf16_kernel<<<(n4w + 255) / 256, 256, 0, stream>>>(Wk, Wkb, n4w);
        cast_bf16_kernel<<<(n4w + 255) / 256, 256, 0, stream>>>(Wv, Wvb, n4w);
        cast_bf16_kernel<<<(n4w + 255) / 256, 256, 0, stream>>>(Wo, Wob, n4w);
    }

    // 2) projections (block tile 128M x 256N)
    dim3 ggrid(D / 256, (unsigned)(M / 128));
    proj_gemm_kernel<<<ggrid, 256, 0, stream>>>(qb, Wqb, bq, Qh, 0);
    proj_gemm_kernel<<<ggrid, 256, 0, stream>>>(kb, Wkb, bk, Kh, 1);
    proj_gemm_kernel<<<ggrid, 256, 0, stream>>>(vb, Wvb, bv, Vt, 2);

    // 3) flash attention: 4 waves/block, each wave = one 64-query block
    dim3 agrid(S / (4 * 64), H, B);
    attn_kernel<<<agrid, 128, 0, stream>>>(Qh, Kh, Vt, AO);

    // 4) output projection (f32 result + bias)
    out_gemm_kernel<<<ggrid, 256, 0, stream>>>(AO, Wob, bo, out);
}
